// FilterModule_47425028883132
// MI455X (gfx1250) — matrix operand order repaired and checked by reading, not compiled, not run on hardware
//
#include <hip/hip_runtime.h>
#include <hip/hip_bf16.h>
#include <math.h>

// ---------------------------------------------------------------------------
// MI455X / gfx1250: wave32, WMMA 16x16x32 bf16 (fp32 accum).
// Whole pipeline is L2-resident (192MB) -> compute bound -> everything that is
// a matmul goes through v_wmma_f32_16x16x32_bf16.
// ---------------------------------------------------------------------------

typedef __attribute__((ext_vector_type(16))) __bf16 v16bf;
typedef __attribute__((ext_vector_type(8)))  float  v8f;

#define WMMA_BF16(a, b, c) \
  __builtin_amdgcn_wmma_f32_16x16x32_bf16(false, (a), false, (b), (short)0, (c), false, false)

static constexpr int kE = 1024;   // embed dim
static constexpr int kH = 4;      // heads
static constexpr int kD = 256;    // head dim
static constexpr int kN = 4096;   // sequence length
static constexpr int kQty = 512;  // 'quantity' (fixed by reference setup)
static constexpr int kB = kN / kQty;

// ---------------------------------------------------------------------------
// Elementwise helpers
// ---------------------------------------------------------------------------
__global__ __launch_bounds__(256) void add_pos_kernel(const float* __restrict__ x,
                                                      const float* __restrict__ pos,
                                                      float* __restrict__ y, int n, int colmask) {
  int i = blockIdx.x * 256 + threadIdx.x;
  if (i < n) y[i] = x[i] + pos[i & colmask];
}

__global__ __launch_bounds__(256) void mul_kernel(const float* __restrict__ a,
                                                  const float* __restrict__ b,
                                                  float* __restrict__ y, int n) {
  int i = blockIdx.x * 256 + threadIdx.x;
  if (i < n) y[i] = a[i] * b[i];
}

__global__ __launch_bounds__(256) void cvt_bf16_kernel(const float* __restrict__ x,
                                                       __bf16* __restrict__ y, int n) {
  int i = blockIdx.x * 256 + threadIdx.x;
  if (i < n) y[i] = (__bf16)x[i];
}

// Transpose + convert: dst[c][r] = (bf16) src[r][c]   (src has leading dim lds_)
__global__ __launch_bounds__(256) void transpose_cvt_kernel(const float* __restrict__ src, int lds_,
                                                            __bf16* __restrict__ dst, int ldd,
                                                            int R, int C) {
  __shared__ float tile[32][33];
  const int bx = blockIdx.x * 32;  // col block
  const int by = blockIdx.y * 32;  // row block
  const int tx = threadIdx.x & 31;
  const int ty = threadIdx.x >> 5;  // 8 row-groups
  (void)R; (void)C;                 // exact multiples by construction
  for (int i = ty; i < 32; i += 8)
    tile[i][tx] = src[(size_t)(by + i) * lds_ + bx + tx];
  __syncthreads();
  for (int i = ty; i < 32; i += 8)
    dst[(size_t)(bx + i) * ldd + by + tx] = (__bf16)tile[tx][i];
}

// Row L2-normalize (norm clamped at 1e-8). Optional f32 and bf16 outputs.
__global__ __launch_bounds__(256) void rownorm_kernel(const float* __restrict__ src,
                                                      float* __restrict__ dstf,
                                                      __bf16* __restrict__ dsth, int cols) {
  const int row = blockIdx.x;
  const int tid = threadIdx.x;
  const float* p = src + (size_t)row * cols;
  float ss = 0.f;
  for (int i = tid; i < cols; i += 256) { float v = p[i]; ss += v * v; }
  for (int off = 16; off; off >>= 1) ss += __shfl_xor(ss, off, 32);
  __shared__ float red[8];
  __shared__ float invn;
  if ((tid & 31) == 0) red[tid >> 5] = ss;
  __syncthreads();
  if (tid == 0) {
    float t = 0.f;
    for (int w = 0; w < 8; ++w) t += red[w];
    invn = 1.f / fmaxf(sqrtf(t), 1e-8f);
  }
  __syncthreads();
  const float inv = invn;
  for (int i = tid; i < cols; i += 256) {
    float v = p[i] * inv;
    if (dstf) dstf[(size_t)row * cols + i] = v;
    if (dsth) dsth[(size_t)row * cols + i] = (__bf16)v;
  }
}

// ---------------------------------------------------------------------------
// WMMA GEMM:  C[M,N] (f32) = A[M,K](f32) . W[N,K](bf16)^T  + bias + residual
// Block tile 64x128, 8 waves (2x4), each wave 32x32 (2x2 WMMA tiles), K step 32.
// A is converted f32->bf16 while staging into LDS. M%64==0, N%128==0, K%32==0.
// Optional batching via blockIdx.z with element strides.
// ---------------------------------------------------------------------------
__global__ __launch_bounds__(256) void gemm_kernel(
    const float* __restrict__ A, int lda, long strideA,
    const __bf16* __restrict__ W, int ldw, long strideW,
    const float* __restrict__ bias,
    const float* __restrict__ Res, int ldr,
    float* __restrict__ C, int ldc, long strideC,
    int K) {
  const int bz = blockIdx.z;
  A += (size_t)bz * strideA;
  W += (size_t)bz * strideW;
  C += (size_t)bz * strideC;
  const int bm = blockIdx.y * 64;
  const int bn = blockIdx.x * 128;
  const int tid = threadIdx.x;
  const int lane = tid & 31;
  const int wave = tid >> 5;
  const int wm = (wave >> 2) * 32;   // 0 / 32
  const int wn = (wave & 3) * 32;    // 0 / 32 / 64 / 96
  const int lrow = lane & 15;
  const int lhi = lane >> 4;

  __shared__ __bf16 As[64][40];     // padded stride: 80B, bank friendly
  __shared__ __bf16 Ws[128][40];

  v8f acc[2][2] = {};

  for (int k0 = 0; k0 < K; k0 += 32) {
    // ---- stage A tile (64x32 f32 -> bf16): 4 threads/row x 8 cols
    {
      const int r = tid >> 2, cs = (tid & 3) * 8;
      const float* p = A + (size_t)(bm + r) * lda + k0 + cs;
      if (k0 + 32 < K) __builtin_prefetch(p + 32, 0, 1);  // global_prefetch path
#pragma unroll
      for (int i = 0; i < 8; ++i) As[r][cs + i] = (__bf16)p[i];
    }
    // ---- stage W tile (128x32 bf16): 2 threads/row x 16 cols
    {
      const int r = tid >> 1, cs = (tid & 1) * 16;
      const __bf16* p = W + (size_t)(bn + r) * ldw + k0 + cs;
      if (k0 + 32 < K) __builtin_prefetch(p + 32, 0, 1);
#pragma unroll
      for (int i = 0; i < 16; ++i) Ws[r][cs + i] = p[i];
    }
    __syncthreads();

    v16bf af[2], bfv[2];
#pragma unroll
    for (int i = 0; i < 2; ++i) {
      const __bf16* p0 = &As[wm + 16 * i + lrow][lhi * 8];
      const __bf16* p1 = &As[wm + 16 * i + lrow][16 + lhi * 8];
#pragma unroll
      for (int e = 0; e < 8; ++e) { af[i][e] = p0[e]; af[i][8 + e] = p1[e]; }
    }
#pragma unroll
    for (int j = 0; j < 2; ++j) {
      const __bf16* p0 = &Ws[wn + 16 * j + lrow][lhi * 8];
      const __bf16* p1 = &Ws[wn + 16 * j + lrow][16 + lhi * 8];
#pragma unroll
      for (int e = 0; e < 8; ++e) { bfv[j][e] = p0[e]; bfv[j][8 + e] = p1[e]; }
    }
#pragma unroll
    for (int i = 0; i < 2; ++i)
#pragma unroll
      for (int j = 0; j < 2; ++j) acc[i][j] = WMMA_BF16(af[i], bfv[j], acc[i][j]);
    __syncthreads();
  }

  // ---- epilogue: bias / residual, f32 store
#pragma unroll
  for (int i = 0; i < 2; ++i)
#pragma unroll
    for (int j = 0; j < 2; ++j) {
      const int gn = bn + wn + 16 * j + lrow;
      const float bv = bias ? bias[gn] : 0.f;
#pragma unroll
      for (int r = 0; r < 8; ++r) {
        const int gm = bm + wm + 16 * i + 8 * lhi + r;
        float v = acc[i][j][r] + bv;
        if (Res) v += Res[(size_t)gm * ldr + gn];
        C[(size_t)gm * ldc + gn] = v;
      }
    }
}

// ---------------------------------------------------------------------------
// Flash attention (one head slice, online softmax).
// Grid: (Lq/64, H). Block 128 = 4 waves, each wave owns 16 query rows.
// qp: [Lq, ldq] f32 (head h at col h*kD), kp: [Lk, ldkv] f32,
// vpT: [E, Lk] bf16 (row h*kD+n = V column n of head h), out: [Lq, ldo] f32.
// ---------------------------------------------------------------------------
__global__ __launch_bounds__(128) void attn_kernel(
    const float* __restrict__ qp, int ldq,
    const float* __restrict__ kp, int ldkv,
    const __bf16* __restrict__ vpT,
    float* __restrict__ out, int ldo, int Lk) {
  const int h = blockIdx.y;
  const int lane = threadIdx.x & 31;
  const int wave = threadIdx.x >> 5;
  const int q0 = blockIdx.x * 64 + wave * 16;
  const int lrow = lane & 15;
  const int lhi = lane >> 4;
  const float scale = 0.0625f;  // 1/sqrt(256)

  __shared__ __bf16 Qs[4][16][kD + 8];
  __shared__ __bf16 Ps[4][16][64 + 8];

  // Stage this wave's Q tile (16 x 256), scaled, f32 -> bf16.
  {
    const float* qbase = qp + (size_t)q0 * ldq + h * kD;
    for (int r = 0; r < 16; ++r) {
      const float* pr = qbase + (size_t)r * ldq + lane * 8;
#pragma unroll
      for (int i = 0; i < 8; ++i) Qs[wave][r][lane * 8 + i] = (__bf16)(pr[i] * scale);
    }
  }

  v8f oacc[16];
#pragma unroll
  for (int t = 0; t < 16; ++t) oacc[t] = v8f{};
  float rmax[8], rsum[8];
#pragma unroll
  for (int r = 0; r < 8; ++r) { rmax[r] = -INFINITY; rsum[r] = 0.f; }

  for (int kt = 0; kt < Lk; kt += 64) {
    // ---- S tile = Q(16x256) . K(64x256)^T, 4 key-subtiles of 16
    v8f sacc[4];
#pragma unroll
    for (int j = 0; j < 4; ++j) sacc[j] = v8f{};
    for (int kk = 0; kk < kD; kk += 32) {
      v16bf a;
      {
        const __bf16* p0 = &Qs[wave][lrow][kk + lhi * 8];
        const __bf16* p1 = &Qs[wave][lrow][kk + 16 + lhi * 8];
#pragma unroll
        for (int e = 0; e < 8; ++e) { a[e] = p0[e]; a[8 + e] = p1[e]; }
      }
#pragma unroll
      for (int j = 0; j < 4; ++j) {
        const float* p = kp + (size_t)(kt + j * 16 + lrow) * ldkv + h * kD + kk + lhi * 8;
        v16bf b;
#pragma unroll
        for (int e = 0; e < 8; ++e) { b[e] = (__bf16)p[e]; b[8 + e] = (__bf16)p[16 + e]; }
        sacc[j] = WMMA_BF16(a, b, sacc[j]);
      }
    }

    // ---- online softmax: row stats via 16-lane shuffles
    float alpha[8], tsum[8];
#pragma unroll
    for (int r = 0; r < 8; ++r) {
      float v = fmaxf(fmaxf(sacc[0][r], sacc[1][r]), fmaxf(sacc[2][r], sacc[3][r]));
      for (int off = 8; off; off >>= 1) v = fmaxf(v, __shfl_xor(v, off, 16));
      const float mnew = fmaxf(rmax[r], v);
      alpha[r] = __expf(rmax[r] - mnew);
      rmax[r] = mnew;
      tsum[r] = 0.f;
    }
#pragma unroll
    for (int j = 0; j < 4; ++j)
#pragma unroll
      for (int r = 0; r < 8; ++r) {
        const float p = __expf(sacc[j][r] - rmax[r]);
        sacc[j][r] = p;
        tsum[r] += p;
      }
#pragma unroll
    for (int r = 0; r < 8; ++r) {
      float s = tsum[r];
      for (int off = 8; off; off >>= 1) s += __shfl_xor(s, off, 16);
      rsum[r] = rsum[r] * alpha[r] + s;
    }
#pragma unroll
    for (int t = 0; t < 16; ++t)
#pragma unroll
      for (int r = 0; r < 8; ++r) oacc[t][r] *= alpha[r];

    // ---- P (bf16) -> wave-private LDS, re-read as A fragments
#pragma unroll
    for (int j = 0; j < 4; ++j)
#pragma unroll
      for (int r = 0; r < 8; ++r)
        Ps[wave][r + 8 * lhi][j * 16 + lrow] = (__bf16)sacc[j][r];

#pragma unroll
    for (int kk2 = 0; kk2 < 64; kk2 += 32) {
      v16bf a;
      {
        const __bf16* p0 = &Ps[wave][lrow][kk2 + lhi * 8];
        const __bf16* p1 = &Ps[wave][lrow][kk2 + 16 + lhi * 8];
#pragma unroll
        for (int e = 0; e < 8; ++e) { a[e] = p0[e]; a[8 + e] = p1[e]; }
      }
#pragma unroll
      for (int t = 0; t < 16; ++t) {
        const __bf16* p = vpT + (size_t)(h * kD + t * 16 + lrow) * Lk + kt + kk2 + lhi * 8;
        v16bf b;
#pragma unroll
        for (int e = 0; e < 8; ++e) { b[e] = p[e]; b[8 + e] = p[16 + e]; }
        oacc[t] = WMMA_BF16(a, b, oacc[t]);
      }
    }
  }

  // ---- finalize: divide by row sums, store f32
#pragma unroll
  for (int r = 0; r < 8; ++r) rsum[r] = 1.f / rsum[r];
#pragma unroll
  for (int t = 0; t < 16; ++t)
#pragma unroll
    for (int r = 0; r < 8; ++r)
      out[(size_t)(q0 + 8 * lhi + r) * ldo + h * kD + t * 16 + lrow] = oacc[t][r] * rsum[r];
}

// ---------------------------------------------------------------------------
// Host-side orchestration
// ---------------------------------------------------------------------------
static void launch_gemm(const float* A, int lda, long sA,
                        const __bf16* W, int ldw, long sW,
                        const float* bias, const float* Res, int ldr,
                        float* C, int ldc, long sC,
                        int M, int N, int K, int nb, hipStream_t s) {
  dim3 grid(N / 128, M / 64, nb);
  gemm_kernel<<<grid, 256, 0, s>>>(A, lda, sA, W, ldw, sW, bias, Res, ldr, C, ldc, sC, K);
}

struct MhaWs {
  float* qproj;    // [N, E]
  float* kvproj;   // [N, 2E]
  __bf16* vpT;     // [E, N]
  float* attnO;    // [N, E]
};

static void run_mha(const float* q_src, const float* kv_src,
                    const __bf16* win, const float* in_b,
                    const __bf16* wout, const float* out_b,
                    const float* residual, float* dest,
                    const MhaWs& ws, hipStream_t s) {
  // Q projection: A=q_src, W = in_w rows [0,E)
  launch_gemm(q_src, kE, 0, win, kE, 0, in_b, nullptr, 0,
              ws.qproj, kE, 0, kN, kE, kE, 1, s);
  // K,V projections: W = in_w rows [E,3E)
  launch_gemm(kv_src, kE, 0, win + (size_t)kE * kE, kE, 0, in_b + kE, nullptr, 0,
              ws.kvproj, 2 * kE, 0, kN, 2 * kE, kE, 1, s);
  // V^T (bf16) for the P.V B-fragments
  {
    dim3 grid(kE / 32, kN / 32);
    transpose_cvt_kernel<<<grid, 256, 0, s>>>(ws.kvproj + kE, 2 * kE, ws.vpT, kN, kN, kE);
  }
  // attention
  {
    dim3 grid(kN / 64, kH);
    attn_kernel<<<grid, 128, 0, s>>>(ws.qproj, kE, ws.kvproj, 2 * kE, ws.vpT, ws.attnO, kE, kN);
  }
  // output projection (+residual)
  launch_gemm(ws.attnO, kE, 0, wout, kE, 0, out_b, residual, kE,
              dest, kE, 0, kN, kE, kE, 1, s);
}

extern "C" void kernel_launch(void* const* d_in, const int* in_sizes, int n_in,
                              void* d_out, int out_size, void* d_ws, size_t ws_size,
                              hipStream_t stream) {
  (void)in_sizes; (void)n_in; (void)out_size; (void)ws_size;
  const float* local_feat = (const float*)d_in[0];
  const float* global_feat = (const float*)d_in[1];
  const float* text_feat = (const float*)d_in[2];
  const float* pos_l = (const float*)d_in[3];
  const float* pos_g = (const float*)d_in[4];
  const float* fc_w = (const float*)d_in[5];
  const float* fc_b = (const float*)d_in[6];
  const float* lg_in_w = (const float*)d_in[7];
  const float* lg_in_b = (const float*)d_in[8];
  const float* lg_out_w = (const float*)d_in[9];
  const float* lg_out_b = (const float*)d_in[10];
  const float* vt_in_w = (const float*)d_in[11];
  const float* vt_in_b = (const float*)d_in[12];
  const float* vt_out_w = (const float*)d_in[13];
  const float* vt_out_b = (const float*)d_in[14];
  const float* ml_in_w = (const float*)d_in[15];
  const float* ml_in_b = (const float*)d_in[16];
  const float* ml_out_w = (const float*)d_in[17];
  const float* ml_out_b = (const float*)d_in[18];
  // d_in[19] = quantity (512, fixed by the reference setup; hardcoded as kQty)

  // ---- carve workspace (~300MB; all intermediates live here)
  char* p = (char*)d_ws;
  auto alloc = [&](size_t bytes) -> void* {
    void* r = (void*)p;
    p += (bytes + 255) & ~(size_t)255;
    return r;
  };
  const size_t MAT = (size_t)kN * kE;  // 4M elems
  __bf16* wfc = (__bf16*)alloc(MAT / 4 * 2);               // [E,E]
  __bf16* wlgi = (__bf16*)alloc(3 * (size_t)kE * kE * 2);
  __bf16* wlgo = (__bf16*)alloc((size_t)kE * kE * 2);
  __bf16* wvti = (__bf16*)alloc(3 * (size_t)kE * kE * 2);
  __bf16* wvto = (__bf16*)alloc((size_t)kE * kE * 2);
  __bf16* wmli = (__bf16*)alloc(3 * (size_t)kE * kE * 2);
  __bf16* wmlo = (__bf16*)alloc((size_t)kE * kE * 2);
  float* lf = (float*)alloc(MAT * 4);
  float* gf = (float*)alloc(MAT * 4);
  float* tf = (float*)alloc(MAT * 4);
  float* m_lg = (float*)alloc(MAT * 4);    // local_global
  float* m_tl = (float*)alloc(MAT * 4);    // text_local
  float* m_tg = (float*)alloc(MAT * 4);    // text_global
  float* m_a = (float*)alloc(MAT * 4);     // gt_gl
  float* m_b = (float*)alloc(MAT * 4);     // gl_lt
  float* m_full = (float*)alloc(MAT * 4);
  float* mtmp = (float*)alloc(MAT * 4);    // mha(tf, full) before multiply
  float* fusion = (float*)alloc(MAT * 4);
  float* ffn = (float*)alloc(MAT * 4);     // l2norm(fusion) f32 (GEMM A side)
  __bf16* lbn = (__bf16*)alloc(MAT * 2);   // l2norm(local_feat) bf16 (GEMM B side)
  MhaWs mws;
  mws.qproj = (float*)alloc(MAT * 4);
  mws.kvproj = (float*)alloc(MAT * 8);
  mws.vpT = (__bf16*)alloc(MAT * 2);
  mws.attnO = (float*)alloc(MAT * 4);

  const int nElem = (int)MAT;
  const int eBlocks = (nElem + 255) / 256;

  // ---- positional adds
  add_pos_kernel<<<eBlocks, 256, 0, stream>>>(local_feat, pos_l, lf, nElem, kE - 1);
  add_pos_kernel<<<eBlocks, 256, 0, stream>>>(global_feat, pos_g, gf, nElem, kE - 1);

  // ---- weights -> bf16
  const int w1 = kE * kE, w3 = 3 * kE * kE;
  cvt_bf16_kernel<<<(w1 + 255) / 256, 256, 0, stream>>>(fc_w, wfc, w1);
  cvt_bf16_kernel<<<(w3 + 255) / 256, 256, 0, stream>>>(lg_in_w, wlgi, w3);
  cvt_bf16_kernel<<<(w1 + 255) / 256, 256, 0, stream>>>(lg_out_w, wlgo, w1);
  cvt_bf16_kernel<<<(w3 + 255) / 256, 256, 0, stream>>>(vt_in_w, wvti, w3);
  cvt_bf16_kernel<<<(w1 + 255) / 256, 256, 0, stream>>>(vt_out_w, wvto, w1);
  cvt_bf16_kernel<<<(w3 + 255) / 256, 256, 0, stream>>>(ml_in_w, wmli, w3);
  cvt_bf16_kernel<<<(w1 + 255) / 256, 256, 0, stream>>>(ml_out_w, wmlo, w1);

  // ---- tf = text_feat @ fc_w.T + fc_b
  launch_gemm(text_feat, kE, 0, wfc, kE, 0, fc_b, nullptr, 0, tf, kE, 0, kN, kE, kE, 1, stream);

  // ---- the seven MHAs
  run_mha(lf, gf, wlgi, lg_in_b, wlgo, lg_out_b, lf, m_lg, mws, stream);      // local_global
  run_mha(tf, lf, wvti, vt_in_b, wvto, vt_out_b, lf, m_tl, mws, stream);      // text_local
  run_mha(tf, gf, wvti, vt_in_b, wvto, vt_out_b, tf, m_tg, mws, stream);      // text_global
  run_mha(m_lg, m_tg, wmli, ml_in_b, wmlo, ml_out_b, nullptr, m_a, mws, stream);  // gt_gl
  run_mha(m_tl, m_lg, wmli, ml_in_b, wmlo, ml_out_b, nullptr, m_b, mws, stream);  // gl_lt
  run_mha(m_b, m_a, wmli, ml_in_b, wmlo, ml_out_b, nullptr, m_full, mws, stream); // full
  run_mha(tf, m_full, wmli, ml_in_b, wmlo, ml_out_b, nullptr, mtmp, mws, stream); // fusion pre

  // ---- fusion = mha(tf, full) * tf
  mul_kernel<<<eBlocks, 256, 0, stream>>>(mtmp, tf, fusion, nElem);

  // ---- l2norm rows
  rownorm_kernel<<<kN, 256, 0, stream>>>(fusion, ffn, nullptr, kE);
  rownorm_kernel<<<kN, 256, 0, stream>>>(local_feat, nullptr, lbn, kE);

  // ---- batched cosine logits: [B, Q, Q] straight into d_out
  launch_gemm(ffn, kE, (long)kQty * kE,
              lbn, kE, (long)kQty * kE,
              nullptr, nullptr, 0,
              (float*)d_out, kQty, (long)kQty * kQty,
              kQty, kQty, kE, kB, stream);
}
